// TransformerBlock_35536559407378
// MI455X (gfx1250) — matrix operand order repaired and checked by reading, not compiled, not run on hardware
//
#include <hip/hip_runtime.h>
#include <hip/hip_bf16.h>
#include <math.h>

// ---------------------------------------------------------------------------
// Transformer block forward for MI455X (gfx1250, wave32, WMMA).
// GEMMs use v_wmma_f32_16x16x32_bf16 (f32 accumulate); f32 operands are
// truncated to bf16 while staging through double-buffered LDS in *fragment
// order*, so every WMMA operand is a packed ds_load_b128 pair. Global loads
// for tile k+1 overlap the WMMAs of tile k; B fragments are register-
// pipelined so LDS latency hides under WMMA issue.
// ---------------------------------------------------------------------------

typedef __attribute__((ext_vector_type(16))) __bf16 v16bf;
typedef __attribute__((ext_vector_type(4)))  __bf16 v4bf;
typedef __attribute__((ext_vector_type(2)))  __bf16 v2bf;
typedef __attribute__((ext_vector_type(8)))  float  v8f;

#define BM 128
#define BK 32
#define NT 256   // 8 wave32 per workgroup

// B-tile LDS layout, fragment order:
//   Bs[((c*16 + n)*2 + khalf)*16 + ki]  ==  B[khalf*16 + ki][c*16 + n]
// lane (n = lane&15, khalf = lane>>4) reads its 16 k-consecutive values for
// column tile c as ONE aligned 32-byte vector (2 x ds_load_b128).
#define BIDX(c, n, h, ki) (((((c) << 4) + (n)) * 2 + (h)) * 16 + (ki))

// ---------------------------------------------------------------------------
// Tiled GEMM: C[M,N] = epilogue(scale * A[M,K] @ B[K,N] + bias)
//  BT=true : B stored [N,K] row-major (Q @ K^T).   EPI==1 : exact GELU.
//  BNT     : number of 16-wide column tiles per block (4 -> BN=64, 8 -> BN=128)
// Batched via blockIdx.z with (outer,inner) offset split (per-(b,h) strides
// of the packed [B,S,H,D] q/k/v layout).
// Requires M%128==0, N%(16*BNT)==0, K%32==0 (true for every GEMM here).
// ---------------------------------------------------------------------------
template<int EPI, bool BT, int BNT>
__global__ void __launch_bounds__(NT)
gemm_bf16_wmma(const float* __restrict__ A, const float* __restrict__ Bm,
               const float* __restrict__ bias, float* __restrict__ C,
               int M, int N, int K, int lda, int ldb, int ldc,
               int zDiv, long aOuter, long aInner, long bOuter, long bInner,
               long cOuter, long cInner, float scale)
{
    __shared__ __bf16 As[2][BM * BK];           // 2 x 8 KB, row-major
    __shared__ __bf16 Bs[2][BK * 16 * BNT];     // 2 x BNT KB, fragment order

    const int tid  = threadIdx.x;
    const int wave = tid >> 5;
    const int lane = tid & 31;

    const int z = blockIdx.z;
    A  += (long)(z / zDiv) * aOuter + (long)(z % zDiv) * aInner;
    Bm += (long)(z / zDiv) * bOuter + (long)(z % zDiv) * bInner;
    C  += (long)(z / zDiv) * cOuter + (long)(z % zDiv) * cInner;

    const int m0 = blockIdx.y * BM;
    const int n0 = blockIdx.x * (16 * BNT);

    // --- tile staging (global f32 -> bf16 LDS), split into a load phase and
    // a store phase so the global_load_b128s pipeline instead of each one
    // stalling at s_wait_loadcnt 0. Prefetches are speculative (dropped on
    // translation fault), so no range guard is needed.
    auto stageA = [&](int k0, int buf) {
        float4 f[4];
        #pragma unroll
        for (int c = 0; c < 4; ++c) {
            const int lin4 = c * NT + tid;     // 0..1023
            const int row  = lin4 >> 3;        // 8 float4 per 32-wide row
            const int col  = (lin4 & 7) << 2;
            const float* gp = A + (long)(m0 + row) * lda + k0 + col;
            __builtin_prefetch(gp + BK, 0, 0);
            f[c] = *(const float4*)gp;
        }
        #pragma unroll
        for (int c = 0; c < 4; ++c) {
            const int lin4 = c * NT + tid;
            const int row  = lin4 >> 3;
            const int col  = (lin4 & 7) << 2;
            *(v4bf*)&As[buf][row * BK + col] =
                (v4bf){(__bf16)f[c].x, (__bf16)f[c].y, (__bf16)f[c].z, (__bf16)f[c].w};
        }
    };
    auto stageB = [&](int k0, int buf) {
        if (BT) {
            // Bm is [N,K] row-major: k-contiguous in global -> v4bf stores.
            float4 f[BNT / 2];
            #pragma unroll
            for (int c = 0; c < BNT / 2; ++c) {
                const int lin4 = c * NT + tid;
                const int n    = lin4 >> 3;    // 8 float4 per 32-k row
                const int kc   = (lin4 & 7) << 2;
                const float* gp = Bm + (long)(n0 + n) * ldb + k0 + kc;
                __builtin_prefetch(gp + BK, 0, 0);
                f[c] = *(const float4*)gp;
            }
            #pragma unroll
            for (int c = 0; c < BNT / 2; ++c) {
                const int lin4 = c * NT + tid;
                const int n    = lin4 >> 3;
                const int kc   = (lin4 & 7) << 2;
                *(v4bf*)&Bs[buf][BIDX(n >> 4, n & 15, kc >> 4, kc & 15)] =
                    (v4bf){(__bf16)f[c].x, (__bf16)f[c].y, (__bf16)f[c].z, (__bf16)f[c].w};
            }
        } else {
            // Bm is [K,N]: each thread stages a 2k x BNT-n micro-tile so the
            // LDS writes are packed k-pairs in fragment order.
            const int kk  = (tid >> 4) << 1;         // 0,2,..,30
            const int nn  = (tid & 15) * BNT;
            const int h   = kk >> 4;
            const int ki0 = kk & 15;
            float4 f0[BNT / 4], f1[BNT / 4];
            #pragma unroll
            for (int q4 = 0; q4 < BNT / 4; ++q4) {
                const float* gp0 = Bm + (long)(k0 + kk) * ldb + n0 + nn + q4 * 4;
                __builtin_prefetch(gp0 + (long)BK * ldb, 0, 0);
                f0[q4] = *(const float4*)gp0;
                f1[q4] = *(const float4*)(gp0 + ldb);
            }
            #pragma unroll
            for (int q4 = 0; q4 < BNT / 4; ++q4) {
                #pragma unroll
                for (int j = 0; j < 4; ++j) {
                    const int ncol = nn + q4 * 4 + j;
                    *(v2bf*)&Bs[buf][BIDX(ncol >> 4, ncol & 15, h, ki0)] =
                        (v2bf){(__bf16)(&f0[q4].x)[j], (__bf16)(&f1[q4].x)[j]};
                }
            }
        }
    };

    v8f acc[BNT] = {};               // wave's 16 x (16*BNT) strip of C

    const int mw   = wave << 4;      // wave's 16-row strip inside block tile
    // A layout (16x32 bf16), matching B fragment order: lanes0-15 row M=lane,
    // K {0..15}; lanes16-31 row M=lane-16, K {16..31}.
    const int am   = lane & 15;
    const int akh  = (lane >> 4) << 4;  // 0 or 16
    const int bn   = lane & 15;         // B fragment column
    const int bh   = lane >> 4;         // B fragment k-half (0 or 1)

    // --- software-pipelined main loop: stage k+1 while computing k ---------
    stageA(0, 0);
    stageB(0, 0);
    int buf = 0;
    for (int k0 = 0; k0 < K; k0 += BK) {
        __syncthreads();             // staging of `buf` complete (and prior
                                     // compute reads of buf^1 retired)
        if (k0 + BK < K) {           // overlap next tile's loads with WMMAs
            stageA(k0 + BK, buf ^ 1);
            stageB(k0 + BK, buf ^ 1);
        }
        // A fragment: two aligned 16B LDS reads, reused by all BNT WMMAs
        v16bf a;
        #pragma unroll
        for (int i = 0; i < 8; ++i) {
            a[i]     = As[buf][(mw + am) * BK + akh + i];
            a[i + 8] = As[buf][(mw + am) * BK + akh + 8 + i];
        }
        // BNT WMMAs; B fragment register-pipelined (load c+1 before WMMA c)
        v16bf bcur = *(const v16bf*)&Bs[buf][BIDX(0, bn, bh, 0)];
        #pragma unroll
        for (int c = 0; c < BNT; ++c) {
            v16bf bnxt;
            if (c + 1 < BNT)
                bnxt = *(const v16bf*)&Bs[buf][BIDX(c + 1, bn, bh, 0)];
            acc[c] = __builtin_amdgcn_wmma_f32_16x16x32_bf16(
                false, a, false, bcur, (short)0, acc[c], false, false);
            if (c + 1 < BNT) bcur = bnxt;
        }
        buf ^= 1;
    }

    // --- epilogue (C layout: VGPR r -> M = r + 8*(lane>=16), N = lane&15)
    const int rowBase = m0 + mw + ((lane >> 4) << 3);
    const int colBase = n0 + (lane & 15);
    #pragma unroll
    for (int c = 0; c < BNT; ++c) {
        #pragma unroll
        for (int r = 0; r < 8; ++r) {
            const int row = rowBase + r;
            const int col = colBase + (c << 4);
            float v = acc[c][r] * scale;
            if (bias) v += bias[col];
            if (EPI == 1)   // exact GELU
                v = 0.5f * v * (1.0f + erff(v * 0.70710678118654752f));
            C[(long)row * ldc + col] = v;
        }
    }
}

// ---------------------------------------------------------------------------
// Row softmax, in place. One 256-thread block per 2048-col row.
// ---------------------------------------------------------------------------
__global__ void __launch_bounds__(NT)
softmax_rows(float* __restrict__ p, int cols)
{
    const int tid = threadIdx.x;
    float* pr = p + (long)blockIdx.x * cols;

    float v[8];
    float m = -INFINITY;
    #pragma unroll
    for (int i = 0; i < 8; ++i) { v[i] = pr[tid + i * NT]; m = fmaxf(m, v[i]); }

    __shared__ float red[NT];
    red[tid] = m; __syncthreads();
    for (int off = NT / 2; off > 0; off >>= 1) {
        if (tid < off) red[tid] = fmaxf(red[tid], red[tid + off]);
        __syncthreads();
    }
    m = red[0]; __syncthreads();

    float s = 0.f;
    #pragma unroll
    for (int i = 0; i < 8; ++i) { v[i] = __expf(v[i] - m); s += v[i]; }
    red[tid] = s; __syncthreads();
    for (int off = NT / 2; off > 0; off >>= 1) {
        if (tid < off) red[tid] += red[tid + off];
        __syncthreads();
    }
    const float inv = 1.0f / red[0];
    #pragma unroll
    for (int i = 0; i < 8; ++i) pr[tid + i * NT] = v[i] * inv;
}

// ---------------------------------------------------------------------------
// out[row] = LayerNorm(x[row] + y[row]) * gamma + beta.  cols == 1024.
// ---------------------------------------------------------------------------
__global__ void __launch_bounds__(NT)
add_layernorm(const float* __restrict__ x, const float* __restrict__ y,
              const float* __restrict__ gamma, const float* __restrict__ beta,
              float* __restrict__ out, int cols)
{
    const int tid = threadIdx.x;
    const long row = blockIdx.x;
    const float* px = x + row * cols;
    const float* py = y + row * cols;

    float v[4], s = 0.f, s2 = 0.f;
    #pragma unroll
    for (int i = 0; i < 4; ++i) {
        const int c = tid + i * NT;
        v[i] = px[c] + py[c];
        s  += v[i];
        s2 += v[i] * v[i];
    }
    __shared__ float rs[NT], rq[NT];
    rs[tid] = s; rq[tid] = s2; __syncthreads();
    for (int off = NT / 2; off > 0; off >>= 1) {
        if (tid < off) { rs[tid] += rs[tid + off]; rq[tid] += rq[tid + off]; }
        __syncthreads();
    }
    const float mean = rs[0] / cols;
    const float var  = rq[0] / cols - mean * mean;
    const float inv  = rsqrtf(var + 1e-5f);
    #pragma unroll
    for (int i = 0; i < 4; ++i) {
        const int c = tid + i * NT;
        out[row * cols + c] = (v[i] - mean) * inv * gamma[c] + beta[c];
    }
}

// ---------------------------------------------------------------------------
extern "C" void kernel_launch(void* const* d_in, const int* in_sizes, int n_in,
                              void* d_out, int out_size, void* d_ws, size_t ws_size,
                              hipStream_t stream)
{
    (void)in_sizes; (void)n_in; (void)out_size; (void)ws_size;

    constexpr int Bt = 2, S = 2048, E = 1024, H = 16, D = 64, HID = 4096;
    constexpr int M = Bt * S;                 // 4096 token rows
    constexpr long TOK = (long)M * E;         // 4,194,304 f32 per activation

    const float* x  = (const float*)d_in[0];
    const float* Wq = (const float*)d_in[1];  const float* bq = (const float*)d_in[2];
    const float* Wk = (const float*)d_in[3];  const float* bk = (const float*)d_in[4];
    const float* Wv = (const float*)d_in[5];  const float* bv = (const float*)d_in[6];
    const float* Wo = (const float*)d_in[7];  const float* bo = (const float*)d_in[8];
    const float* W1 = (const float*)d_in[9];  const float* b1 = (const float*)d_in[10];
    const float* W2 = (const float*)d_in[11]; const float* b2 = (const float*)d_in[12];
    const float* gamma = (const float*)d_in[13];
    const float* beta  = (const float*)d_in[14];

    // Workspace layout (f32). q/k/v/attnout region is reused as mlp1 once h
    // is formed; proj slot is reused as mlp2. Peak: 25,165,824 f32 (~100 MB).
    float* ws      = (float*)d_ws;
    float* q       = ws;                 // TOK
    float* k       = ws + 1 * TOK;
    float* v       = ws + 2 * TOK;
    float* attnout = ws + 3 * TOK;
    float* proj    = ws + 4 * TOK;       // later: mlp2
    float* h       = ws + 5 * TOK;
    float* mlp1    = ws;                 // 4*TOK f32, overlays q..attnout
    float* mlp2    = proj;

    float* out  = (float*)d_out;         // [B,S,E]
    float* attn = out + TOK;             // [B,H,S,S] (pre-softmax energy first)

    const dim3 blk(NT);
    const long zero = 0;

    // --- QKV projections: [4096,1024] @ [1024,1024] + bias   (BN=128)
    const dim3 gProj(E / 128, M / BM, 1);
    gemm_bf16_wmma<0, false, 8><<<gProj, blk, 0, stream>>>(
        x, Wq, bq, q, M, E, E, E, E, E, 1, zero, zero, zero, zero, zero, zero, 1.0f);
    gemm_bf16_wmma<0, false, 8><<<gProj, blk, 0, stream>>>(
        x, Wk, bk, k, M, E, E, E, E, E, 1, zero, zero, zero, zero, zero, zero, 1.0f);
    gemm_bf16_wmma<0, false, 8><<<gProj, blk, 0, stream>>>(
        x, Wv, bv, v, M, E, E, E, E, E, 1, zero, zero, zero, zero, zero, zero, 1.0f);

    // --- energy = (Q @ K^T) / sqrt(D), per (b,h). q/k rows stride E, head off 64.
    const dim3 gEn(S / 128, S / BM, Bt * H);
    gemm_bf16_wmma<0, true, 8><<<gEn, blk, 0, stream>>>(
        q, k, nullptr, attn, S, S, D, E, E, S,
        H, (long)S * E, (long)D, (long)S * E, (long)D,
        (long)H * S * S, (long)S * S, 0.125f);

    // --- softmax over last axis, in place (B*H*S rows of S)
    softmax_rows<<<dim3(Bt * H * S), blk, 0, stream>>>(attn, S);

    // --- attn_out = attn @ V  (per head; N=64 -> BNT=4; packed [B,S,H,D])
    const dim3 gAV(D / 64, S / BM, Bt * H);
    gemm_bf16_wmma<0, false, 4><<<gAV, blk, 0, stream>>>(
        attn, v, nullptr, attnout, S, D, S, S, E, E,
        H, (long)H * S * S, (long)S * S, (long)S * E, (long)D,
        (long)S * E, (long)D, 1.0f);

    // --- output projection
    gemm_bf16_wmma<0, false, 8><<<gProj, blk, 0, stream>>>(
        attnout, Wo, bo, proj, M, E, E, E, E, E,
        1, zero, zero, zero, zero, zero, zero, 1.0f);

    // --- h = LN(x + proj)
    add_layernorm<<<dim3(M), blk, 0, stream>>>(x, proj, gamma, beta, h, E);

    // --- mlp1 = gelu(h @ W1 + b1)   [4096,4096]
    const dim3 gFF1(HID / 128, M / BM, 1);
    gemm_bf16_wmma<1, false, 8><<<gFF1, blk, 0, stream>>>(
        h, W1, b1, mlp1, M, HID, E, E, HID, HID,
        1, zero, zero, zero, zero, zero, zero, 1.0f);

    // --- mlp2 = mlp1 @ W2 + b2
    gemm_bf16_wmma<0, false, 8><<<gProj, blk, 0, stream>>>(
        mlp1, W2, b2, mlp2, M, E, HID, HID, E, E,
        1, zero, zero, zero, zero, zero, zero, 1.0f);

    // --- out = LN(h + mlp2)
    add_layernorm<<<dim3(M), blk, 0, stream>>>(h, mlp2, gamma, beta, out, E);
}
